// GCN2ConvEnsemble_83133386981994
// MI455X (gfx1250) — matrix operand order repaired and checked by reading, compiled, not run on hardware
//
#include <hip/hip_runtime.h>
#include <hip/hip_bf16.h>
#include <math.h>

// ---------------- problem constants (match reference) ----------------
#define NN    100000
#define EE    1600000
#define INF_  128
#define HF    128
#define OUTF  112
#define LL    4
#define NETS_ 3
#define ALPHA_C 0.1f

typedef __attribute__((ext_vector_type(16))) __bf16 v16bf;
typedef __attribute__((ext_vector_type(8)))  float  v8f;

// ---------------------------------------------------------------------
// WMMA bf16 GEMM:  D = epilogue( A[Mx128] @ B[128xNCOLS] ), K fixed at 128.
//   block = (NCOLS/16) waves, each wave owns one 16-wide column tile
//   grid.x = M/16 row tiles (M = 100000 = 6250*16 exactly)
// MODE 0: D = relu(acc + bias[col]);  D2 = D       (lin0 -> h and x0)
// MODE 1: D = relu(c0 * T[row,col] + c1 * acc)     (GCN2 layer, scales folded)
// MODE 2: D = acc + bias[col]                      (lin1 -> logits)
// NCOLS compile-time => all B/T/D strides become immediate offsets.
// ---------------------------------------------------------------------
template <int MODE, int NCOLS>
__global__ __launch_bounds__(256)
void gemm_wmma_kernel(const float* __restrict__ A,
                      const float* __restrict__ B,
                      const float* __restrict__ bias,
                      const float* __restrict__ T,
                      float* __restrict__ D,
                      float* __restrict__ D2,
                      float c0, float c1)
{
    const int wave = threadIdx.x >> 5;
    const int lane = threadIdx.x & 31;
    const int g    = lane >> 4;    // lane half
    const int mr   = lane & 15;    // row (A) / col (B, C/D) within tile
    const int m0   = blockIdx.x * 16;
    const int n0   = wave * 16;
    const int col  = n0 + mr;

    const float* __restrict__ Arow  = A + (size_t)(m0 + mr) * HF;
    const float* __restrict__ Bcol  = B + col;          // + kB*NCOLS as immediates

    v8f acc = {};
#pragma unroll
    for (int k0 = 0; k0 < HF; k0 += 32) {
        v16bf a, b;
#pragma unroll
        for (int v = 0; v < 8; ++v) {
            // A 16x32 bf16 layout: K = 16*(v/4) + 8*g + 2*(v%4)
            const int kA = k0 + ((v >> 2) << 4) + (g << 3) + ((v & 3) << 1);
            a[2 * v]     = (__bf16)Arow[kA];
            a[2 * v + 1] = (__bf16)Arow[kA + 1];
            // B 32x16 bf16 layout: K = 16*(v/4) + 8*g + 2*(v%4)
            const int kB = k0 + ((v >> 2) << 4) + (g << 3) + ((v & 3) << 1);
            b[2 * v]     = (__bf16)Bcol[(size_t)kB * NCOLS];
            b[2 * v + 1] = (__bf16)Bcol[(size_t)(kB + 1) * NCOLS];
        }
        acc = __builtin_amdgcn_wmma_f32_16x16x32_bf16(
                  false, a, false, b, (short)0, acc, false, false);
    }

    const float bv = (MODE == 1) ? 0.0f : bias[col];
    const size_t base = (size_t)(m0 + (g << 3)) * NCOLS + col;  // row m0+8g, col
#pragma unroll
    for (int r = 0; r < 8; ++r) {
        // C/D layout: VGPR r -> M = r + 8*g
        const size_t i = base + (size_t)r * NCOLS;
        if (MODE == 0) {
            const float val = fmaxf(acc[r] + bv, 0.0f);
            D[i]  = val;
            D2[i] = val;
        } else if (MODE == 1) {
            D[i] = fmaxf(c0 * T[i] + c1 * acc[r], 0.0f);
        } else {
            D[i] = acc[r] + bv;
        }
    }
}

// ---------------- sparse propagation: agg[dst] += ew * h[src] ----------------
// 32 lanes per edge, 4 features per lane (H = 128); 8 edges per block.
__global__ __launch_bounds__(256)
void spmm_kernel(const int* __restrict__ src, const int* __restrict__ dst,
                 const float* __restrict__ ew, const float* __restrict__ h,
                 float* __restrict__ agg)
{
    const int e = blockIdx.x * 8 + (threadIdx.x >> 5);
    const int f = (threadIdx.x & 31) * 4;
    const int s = src[e];
    const int d = dst[e];
    const float w = ew[e];
    const float4 hv = *(const float4*)(h + (size_t)s * HF + f);
    float* ap = agg + (size_t)d * HF + f;
    unsafeAtomicAdd(ap + 0, w * hv.x);
    unsafeAtomicAdd(ap + 1, w * hv.y);
    unsafeAtomicAdd(ap + 2, w * hv.z);
    unsafeAtomicAdd(ap + 3, w * hv.w);
}

// ---------------- dst = s * src (prefill agg with alpha-residual) ----------------
__global__ void scale_copy_kernel(float* __restrict__ dst, const float* __restrict__ srcv,
                                  size_t n, float s)
{
    size_t i = (size_t)blockIdx.x * blockDim.x + threadIdx.x;
    size_t stride = (size_t)gridDim.x * blockDim.x;
    for (; i < n; i += stride) dst[i] = s * srcv[i];
}

__global__ void zero_kernel(float* __restrict__ p, size_t n)
{
    size_t i = (size_t)blockIdx.x * blockDim.x + threadIdx.x;
    size_t stride = (size_t)gridDim.x * blockDim.x;
    for (; i < n; i += stride) p[i] = 0.0f;
}

// ---------------- out += wscale * log_softmax(logits[node,:]) ----------------
__global__ __launch_bounds__(128)
void logsoftmax_accum_kernel(const float* __restrict__ logits,
                             float* __restrict__ out, float wscale)
{
    const int node = blockIdx.x;
    const int t = threadIdx.x;
    __shared__ float sdata[128];

    const float v = (t < OUTF) ? logits[(size_t)node * OUTF + t] : -__builtin_inff();
    sdata[t] = v;
    __syncthreads();
#pragma unroll
    for (int s = 64; s > 0; s >>= 1) {
        if (t < s) sdata[t] = fmaxf(sdata[t], sdata[t + s]);
        __syncthreads();
    }
    const float m = sdata[0];
    __syncthreads();
    sdata[t] = (t < OUTF) ? expf(v - m) : 0.0f;
    __syncthreads();
#pragma unroll
    for (int s = 64; s > 0; s >>= 1) {
        if (t < s) sdata[t] += sdata[t + s];
        __syncthreads();
    }
    const float lse = logf(sdata[0]) + m;
    if (t < OUTF) out[(size_t)node * OUTF + t] += wscale * (v - lse);
}

// ---------------------------------------------------------------------
extern "C" void kernel_launch(void* const* d_in, const int* in_sizes, int n_in,
                              void* d_out, int out_size, void* d_ws, size_t ws_size,
                              hipStream_t stream)
{
    (void)in_sizes; (void)n_in; (void)out_size; (void)ws_size;

    const float* x      = (const float*)d_in[0];
    const int*   eidx   = (const int*)d_in[1];      // [2, E] flat
    const float* ew     = (const float*)d_in[2];
    const float* lin0_w = (const float*)d_in[3];    // [NETS, IN, H]
    const float* lin0_b = (const float*)d_in[4];    // [NETS, H]
    const float* lin1_w = (const float*)d_in[5];    // [NETS, H, OUT]
    const float* lin1_b = (const float*)d_in[6];    // [NETS, OUT]
    const float* conv_w = (const float*)d_in[7];    // [NETS, L, H, H]
    float* out = (float*)d_out;

    const int* src = eidx;
    const int* dst = eidx + EE;

    // workspace: h | x0 | agg(=t=logits)   — 3 * N*H floats = 153.6 MB
    float* h   = (float*)d_ws;
    float* x0  = h  + (size_t)NN * HF;
    float* agg = x0 + (size_t)NN * HF;

    const size_t nh = (size_t)NN * HF;
    const dim3 zgrid(2048), zblk(256);
    const int mtiles = NN / 16;                      // 6250

    // d_out accumulates the ensemble mean: zero it first.
    zero_kernel<<<zgrid, zblk, 0, stream>>>(out, (size_t)NN * OUTF);

    for (int n = 0; n < NETS_; ++n) {
        // h = relu(x @ w0 + b0); x0 = h
        gemm_wmma_kernel<0, HF><<<mtiles, 32 * (HF / 16), 0, stream>>>(
            x, lin0_w + (size_t)n * INF_ * HF, lin0_b + (size_t)n * HF,
            nullptr, h, x0, 0.f, 0.f);

        for (int l = 0; l < LL; ++l) {
            const float beta = logf(0.5f / (float)(l + 1) + 1.0f);

            // agg := (alpha/(1-alpha)) * x0, then atomics accumulate the SpMM on top;
            // the (1-alpha) factor is folded into the GEMM epilogue coefficients.
            scale_copy_kernel<<<zgrid, zblk, 0, stream>>>(
                agg, x0, nh, ALPHA_C / (1.0f - ALPHA_C));
            spmm_kernel<<<EE / 8, 256, 0, stream>>>(src, dst, ew, h, agg);

            // h = relu((1-beta)*t + beta*(t @ cw[l])),  t = (1-alpha)*agg
            const float c0 = (1.0f - beta) * (1.0f - ALPHA_C);
            const float c1 = beta * (1.0f - ALPHA_C);
            gemm_wmma_kernel<1, HF><<<mtiles, 32 * (HF / 16), 0, stream>>>(
                agg, conv_w + ((size_t)n * LL + l) * HF * HF, nullptr,
                agg, h, nullptr, c0, c1);
        }

        // logits (reuse agg buffer) = h @ w1 + b1
        gemm_wmma_kernel<2, OUTF><<<mtiles, 32 * (OUTF / 16), 0, stream>>>(
            h, lin1_w + (size_t)n * HF * OUTF, lin1_b + (size_t)n * OUTF,
            nullptr, agg, nullptr, 0.f, 0.f);

        // out += (1/3) * log_softmax(logits)
        logsoftmax_accum_kernel<<<NN, 128, 0, stream>>>(agg, out, 1.0f / (float)NETS_);
    }
}
